// Decoder_26620207300925
// MI455X (gfx1250) — matrix operand order repaired and checked by reading, not compiled, not run on hardware
//
#include <hip/hip_runtime.h>
#include <hip/hip_bf16.h>
#include <math.h>

// ---------------------------------------------------------------------------
// Tacotron2-style decoder for MI455X (gfx1250, wave32).
// All heavy GEMMs run on v_wmma_f32_16x16x32_bf16 with fp32 accumulate.
// Weights are converted once per launch into the exact WMMA B-fragment
// register layout (tile-major [n_tile][k_tile][lane][16]) so every operand
// fragment is 32 contiguous bytes per lane -> 2x global_load_b128.
// Each wave now produces TWO 16x16 output tiles (two independent WMMA
// accumulator chains, shared operand fetched once). The ~31MB bf16 weight
// set stays L2-resident across the 250-step sequential scan.
// ---------------------------------------------------------------------------

typedef __attribute__((ext_vector_type(16))) __bf16 v16bf;
typedef __attribute__((ext_vector_type(8)))  float  v8f;

#define BDIM   4
#define TDIM   256
#define LSTEPS 250
#define ODIM   80
#define DUNITS 1024
#define ATTD   256
#define XSK    768   // att_c(512) + prenet(256)

__device__ inline v8f wmma_bf16(v16bf a, v16bf b, v8f c) {
  return __builtin_amdgcn_wmma_f32_16x16x32_bf16(false, a, false, b, (short)0, c,
                                                 false, false);
}

// Fragment load: tile stored as [lane][16 elems] = 32B/lane, contiguous.
__device__ inline v16bf load_frag(const __bf16* tile_base) {
  return *(const v16bf*)(tile_base + (size_t)(threadIdx.x & 31) * 16);
}

// A-fragment swizzle offset for element (row m, col k) of a 16 x K matrix.
// ISA 7.12.2 16-bit A 16x32: lanes 0-15 hold K{0..7,16..23}, lanes 16-31 hold
// K{8..15,24..31}; elem i<8 -> K=kb+i, i>=8 -> K=16+kb+(i-8).
__device__ __host__ inline int a_swz_off(int m, int k) {
  int kk   = k & 31;
  int lane = (((kk >> 3) & 1) << 4) | (m & 15);
  int i    = (kk & 7) | (((kk >> 4) & 1) << 3);
  return ((k >> 5) << 9) + (lane << 4) + i;   // tile*512 + lane*16 + i
}

__device__ inline float sigmoidf(float x) { return 1.0f / (1.0f + expf(-x)); }

// ---------------------------------------------------------------------------
// init: zero states + swizzled A buffers; att_prev = mask/hlen
// ---------------------------------------------------------------------------
__global__ void k_init(float* h0, float* c0, float* h1, float* c1,
                       __bf16* xs_swz, __bf16* h0_swz, __bf16* h1_swz,
                       float* att_prev, const int* __restrict__ hlens) {
  int idx = blockIdx.x * 256 + threadIdx.x;
  if (idx < BDIM * DUNITS) { h0[idx] = 0.f; c0[idx] = 0.f; h1[idx] = 0.f; c1[idx] = 0.f; }
  if (idx < 16 * XSK)      xs_swz[idx] = (__bf16)0.f;
  if (idx < 16 * DUNITS) { h0_swz[idx] = (__bf16)0.f; h1_swz[idx] = (__bf16)0.f; }
  if (idx < BDIM * TDIM) {
    int b = idx >> 8, t = idx & 255;
    int hl = hlens[b];
    att_prev[idx] = (t < hl) ? 1.0f / (float)hl : 0.f;
  }
}

// ---------------------------------------------------------------------------
// Weight swizzlers: fp32 -> bf16 in B-fragment tile-major layout
//   dst[((nt*(K/32)+kt)*32 + lane)*16 + i], value = W[row n, col k]
//   B frag: lanes 0-15 K{0..7,16..23}, lanes 16-31 K{8..15,24..31}, lane%16 = column n.
// ---------------------------------------------------------------------------
// W is (N,K) row-major (torch Linear weight); B operand is W^T (KxN).
__global__ void k_wswz_nk(const float* __restrict__ W, __bf16* Wt, int N, int K) {
  size_t idx = (size_t)blockIdx.x * 256 + threadIdx.x;
  if (idx >= (size_t)N * K) return;
  int i  = (int)(idx & 15);
  int l  = (int)((idx >> 4) & 31);
  int kt = (int)((idx >> 9) % (size_t)(K >> 5));
  int nt = (int)(idx / ((size_t)(K >> 5) << 9));
  int n  = nt * 16 + (l & 15);
  int k  = kt * 32 + ((i < 8) ? (((l & 16) ? 8 : 0) + i) : (16 + ((l & 16) ? 8 : 0) + (i - 8)));
  Wt[idx] = (__bf16)W[(size_t)n * K + k];
}

// W is (K,N) row-major (e.g. W_enc)
__global__ void k_wswz_kn(const float* __restrict__ W, __bf16* Wt, int N, int K) {
  size_t idx = (size_t)blockIdx.x * 256 + threadIdx.x;
  if (idx >= (size_t)N * K) return;
  int i  = (int)(idx & 15);
  int l  = (int)((idx >> 4) & 31);
  int kt = (int)((idx >> 9) % (size_t)(K >> 5));
  int nt = (int)(idx / ((size_t)(K >> 5) << 9));
  int n  = nt * 16 + (l & 15);
  int k  = kt * 32 + ((i < 8) ? (((l & 16) ? 8 : 0) + i) : (16 + ((l & 16) ? 8 : 0) + (i - 8)));
  Wt[idx] = (__bf16)W[(size_t)k * N + n];
}

// postnet conv (Cout,Cin,5) -> B-fragment layout over K = Cin*5, zero-padded
// to Kpad = ceil32(K) so the GEMM loop needs no K guard.
__global__ void k_pwswz(const float* __restrict__ conv, __bf16* Wt, int Cout, int Cin) {
  int K = Cin * 5;
  int Kpad = (K + 31) & ~31;
  size_t total = (size_t)(Cout >> 4) * (Kpad >> 5) * 512;
  size_t idx = (size_t)blockIdx.x * 256 + threadIdx.x;
  if (idx >= total) return;
  int i  = (int)(idx & 15);
  int l  = (int)((idx >> 4) & 31);
  int kt = (int)((idx >> 9) % (size_t)(Kpad >> 5));
  int nt = (int)(idx / ((size_t)(Kpad >> 5) << 9));
  int co = nt * 16 + (l & 15);
  int k  = kt * 32 + ((i < 8) ? (((l & 16) ? 8 : 0) + i) : (16 + ((l & 16) ? 8 : 0) + (i - 8)));
  float v = 0.f;
  if (k < K) v = conv[((size_t)co * Cin + k / 5) * 5 + (k % 5)];
  Wt[idx] = (__bf16)v;
}

// hs (1024,512) fp32 -> A-fragment layout per 16-row tile:
//   dst[((mt*16+kt)*32 + lane)*16 + i]
__global__ void k_hs_swz(const float* __restrict__ hs, __bf16* dst) {
  int idx = blockIdx.x * 256 + threadIdx.x;   // 524288 total
  if (idx >= 1024 * 512) return;
  int i  = idx & 15;
  int l  = (idx >> 4) & 31;
  int kt = (idx >> 9) & 15;          // 512/32 = 16 k-tiles
  int mt = idx >> 13;
  int m  = mt * 16 + (l & 15);
  int kb = (l & 16) ? 8 : 0;
  int k  = kt * 32 + ((i < 8) ? (kb + i) : (16 + kb + (i - 8)));
  dst[idx] = (__bf16)hs[(size_t)m * 512 + k];
}

// ---------------------------------------------------------------------------
// pre_enc = hs @ W_enc + b_enc : M=1024, K=512, N=256 (WMMA, frag-swizzled)
// grid (32,16) x 32 threads; 2 m-tiles per wave sharing the B fragment.
// ---------------------------------------------------------------------------
__global__ void k_preenc(const __bf16* __restrict__ hs_swz,
                         const __bf16* __restrict__ wenc_swz,
                         const float* __restrict__ b_enc, float* __restrict__ pre_enc) {
  int mt = blockIdx.x * 2, nt = blockIdx.y;
  v8f acc0 = {0.f, 0.f, 0.f, 0.f, 0.f, 0.f, 0.f, 0.f};
  v8f acc1 = acc0;
#pragma unroll 4
  for (int kt = 0; kt < 16; ++kt) {
    v16bf bfrag = load_frag(wenc_swz + ((size_t)nt * 16 + kt) * 512);
    acc0 = wmma_bf16(load_frag(hs_swz + ((size_t)mt * 16 + kt) * 512), bfrag, acc0);
    acc1 = wmma_bf16(load_frag(hs_swz + ((size_t)(mt + 1) * 16 + kt) * 512), bfrag, acc1);
  }
  int lane = threadIdx.x & 31;
  int n = nt * 16 + (lane & 15);
  int mb = (lane & 16) ? 8 : 0;
  float bias = b_enc[n];
#pragma unroll
  for (int i = 0; i < 8; ++i) {
    pre_enc[(size_t)(mt * 16 + mb + i) * 256 + n]       = acc0[i] + bias;
    pre_enc[(size_t)((mt + 1) * 16 + mb + i) * 256 + n] = acc1[i] + bias;
  }
}

// ---------------------------------------------------------------------------
// LSTM gate GEMM: gates(4x4096) = A1@W1 + A2@W2 + b1 + b2
// One wave per TWO 16-wide column tiles (shared A fragment, two independent
// WMMA chains); KT = K/32 compile-time; weight stream prefetched.
// grid 128 x 32 threads.
// ---------------------------------------------------------------------------
template <int KT1, int KT2>
__global__ void k_gates(const __bf16* __restrict__ A1, const __bf16* __restrict__ W1,
                        const __bf16* __restrict__ A2, const __bf16* __restrict__ W2,
                        const float* __restrict__ bias1, const float* __restrict__ bias2,
                        float* __restrict__ gates, int N) {
  int nt = blockIdx.x * 2;
  const __bf16* w1a = W1 + (size_t)nt * KT1 * 512;
  const __bf16* w1b = w1a + (size_t)KT1 * 512;
  const __bf16* w2a = W2 + (size_t)nt * KT2 * 512;
  const __bf16* w2b = w2a + (size_t)KT2 * 512;
  v8f acc0 = {0.f, 0.f, 0.f, 0.f, 0.f, 0.f, 0.f, 0.f};
  v8f acc1 = acc0;
#pragma unroll 4
  for (int kt = 0; kt < KT1; ++kt) {
    __builtin_prefetch(w1a + (size_t)(kt + 8) * 512, 0, 1);
    __builtin_prefetch(w1b + (size_t)(kt + 8) * 512, 0, 1);
    v16bf a = load_frag(A1 + (size_t)kt * 512);
    acc0 = wmma_bf16(a, load_frag(w1a + (size_t)kt * 512), acc0);
    acc1 = wmma_bf16(a, load_frag(w1b + (size_t)kt * 512), acc1);
  }
#pragma unroll 4
  for (int kt = 0; kt < KT2; ++kt) {
    __builtin_prefetch(w2a + (size_t)(kt + 8) * 512, 0, 1);
    __builtin_prefetch(w2b + (size_t)(kt + 8) * 512, 0, 1);
    v16bf a = load_frag(A2 + (size_t)kt * 512);
    acc0 = wmma_bf16(a, load_frag(w2a + (size_t)kt * 512), acc0);
    acc1 = wmma_bf16(a, load_frag(w2b + (size_t)kt * 512), acc1);
  }
  int lane = threadIdx.x & 31;
  int n0 = nt * 16 + (lane & 15);
  int n1 = n0 + 16;
  int mb = (lane & 16) ? 8 : 0;
  float bias0 = bias1[n0] + bias2[n0];
  float bias1v = bias1[n1] + bias2[n1];
#pragma unroll
  for (int i = 0; i < 8; ++i) {
    int m = mb + i;
    if (m < BDIM) {
      gates[(size_t)m * N + n0] = acc0[i] + bias0;
      gates[(size_t)m * N + n1] = acc1[i] + bias1v;
    }
  }
}

// gate nonlinearity + zoneout; writes h fp32 and A-fragment bf16 copy
__global__ void k_lstm_update(const float* __restrict__ gates, float* h, float* c,
                              __bf16* h_swz) {
  int idx = blockIdx.x * blockDim.x + threadIdx.x;   // 4096
  int b = idx >> 10, j = idx & 1023;
  const float* g = gates + (size_t)b * 4096;
  float ig = sigmoidf(g[j]);
  float fg = sigmoidf(g[1024 + j]);
  float gg = tanhf(g[2048 + j]);
  float og = sigmoidf(g[3072 + j]);
  float cn = fg * c[idx] + ig * gg;
  float hn = og * tanhf(cn);
  float hz = 0.1f * h[idx] + 0.9f * hn;
  float cz = 0.1f * c[idx] + 0.9f * cn;
  h[idx] = hz; c[idx] = cz;
  h_swz[a_swz_off(b, j)] = (__bf16)hz;
}

// ---------------------------------------------------------------------------
// Location-sensitive attention, one workgroup (256 thr) per batch item.
// ---------------------------------------------------------------------------
__global__ void k_attention(const float* __restrict__ hs, const int* __restrict__ hlens,
                            const float* __restrict__ trans, const float* __restrict__ pre_enc,
                            const float* __restrict__ h0,
                            const float* __restrict__ W_dec, const float* __restrict__ W_att,
                            const float* __restrict__ loc_conv, const float* __restrict__ gvec_w,
                            const float* __restrict__ gvec_b,
                            float* att_prev, float* att_c, __bf16* xs_swz,
                            float* attw_out, int step) {
  const int b = blockIdx.x;
  const int t = threadIdx.x;
  __shared__ float s_prev[TDIM];
  __shared__ float s_conv[32 * TDIM];
  __shared__ float s_h0[DUNITS];
  __shared__ float s_df[ATTD];
  __shared__ float s_lc[32 * 31];
  __shared__ float s_red[TDIM];
  __shared__ float s_w[TDIM];

  s_prev[t] = att_prev[b * TDIM + t];
  for (int i = t; i < DUNITS; i += TDIM) s_h0[i] = h0[b * DUNITS + i];
  for (int i = t; i < 32 * 31; i += TDIM) s_lc[i] = loc_conv[i];
  __syncthreads();

  for (int cch = 0; cch < 32; ++cch) {
    float acc = 0.f;
#pragma unroll
    for (int j = 0; j < 31; ++j) {
      int tt = t + j - 15;
      float pv = (tt >= 0 && tt < TDIM) ? s_prev[tt] : 0.f;
      acc += s_lc[cch * 31 + j] * pv;
    }
    s_conv[cch * TDIM + t] = acc;
  }
  {
    float acc = 0.f;
    for (int k = 0; k < DUNITS; ++k) acc += s_h0[k] * W_dec[(size_t)k * ATTD + t];
    s_df[t] = acc;
  }
  __syncthreads();

  float conv_t[32];
#pragma unroll
  for (int cch = 0; cch < 32; ++cch) conv_t[cch] = s_conv[cch * TDIM + t];

  float e = 0.f;
  for (int a = 0; a < ATTD; ++a) {
    float af = s_df[a] + pre_enc[((size_t)b * TDIM + t) * ATTD + a];
#pragma unroll
    for (int cch = 0; cch < 32; ++cch) af += conv_t[cch] * W_att[cch * ATTD + a];
    e += gvec_w[a] * tanhf(af);
  }
  e += gvec_b[0];
  int hl = hlens[b];
  bool valid = (t < hl);
  if (!valid) e = -INFINITY;

  s_red[t] = e; __syncthreads();
  for (int s = 128; s > 0; s >>= 1) {
    if (t < s) s_red[t] = fmaxf(s_red[t], s_red[t + s]);
    __syncthreads();
  }
  float emax = s_red[0]; __syncthreads();
  float p = valid ? expf(2.0f * (e - emax)) : 0.f;   // SCALING=2
  s_red[t] = p; __syncthreads();
  for (int s = 128; s > 0; s >>= 1) {
    if (t < s) s_red[t] += s_red[t + s];
    __syncthreads();
  }
  float psum = s_red[0]; __syncthreads();
  float wsm = p / psum;

  s_red[t] = s_prev[t] * trans[b * TDIM + t]; __syncthreads();
  for (int s = 128; s > 0; s >>= 1) {
    if (t < s) s_red[t] += s_red[t + s];
    __syncthreads();
  }
  float g = s_red[0]; __syncthreads();

  float w = g * wsm + (1.f - g) * s_prev[t];
  s_w[t] = w;
  attw_out[((size_t)b * LSTEPS + step) * TDIM + t] = w;
  att_prev[b * TDIM + t] = w;
  __syncthreads();

  for (int d = t; d < 512; d += TDIM) {
    float acc = 0.f;
    for (int tt = 0; tt < TDIM; ++tt)
      acc += s_w[tt] * hs[((size_t)b * TDIM + tt) * 512 + d];
    att_c[b * 512 + d] = acc;
    xs_swz[a_swz_off(b, d)] = (__bf16)acc;   // rows m=4..15 stay zero
  }
}

// ---------------------------------------------------------------------------
// prenet: prev_out(4x80) -> relu(256) -> relu(256), into xs cols 512..767
// ---------------------------------------------------------------------------
__global__ void k_prenet(const float* __restrict__ ys, int step,
                         const float* __restrict__ W1, const float* __restrict__ b1,
                         const float* __restrict__ W2, const float* __restrict__ b2,
                         __bf16* xs_swz) {
  __shared__ float l1[BDIM][256];
  int j = threadIdx.x;
  for (int b = 0; b < BDIM; ++b) {
    float acc = b1[j];
    if (step > 0) {
      const float* y = ys + ((size_t)b * LSTEPS + (step - 1)) * ODIM;
      for (int i = 0; i < ODIM; ++i) acc += y[i] * W1[i * 256 + j];
    }
    l1[b][j] = fmaxf(acc, 0.f);
  }
  __syncthreads();
  for (int b = 0; b < BDIM; ++b) {
    float acc = b2[j];
    for (int i = 0; i < 256; ++i) acc += l1[b][i] * W2[i * 256 + j];
    xs_swz[a_swz_off(b, 512 + j)] = (__bf16)fmaxf(acc, 0.f);
  }
}

// ---------------------------------------------------------------------------
// output head: zcs=[h1,att_c](4x1536) -> before(80), logit(1); also writes
// bf16 postnet input x0[(b*80+o)*250+t]
// ---------------------------------------------------------------------------
__global__ void k_output(const float* __restrict__ h1, const float* __restrict__ att_c,
                         const float* __restrict__ feat_w, const float* __restrict__ prob_w,
                         const float* __restrict__ prob_b,
                         float* before_out, float* logits, __bf16* x0_bf, int step) {
  __shared__ float zcs[BDIM][1536];
  int tid = threadIdx.x;   // 512
  for (int idx = tid; idx < BDIM * 1536; idx += 512) {
    int b = idx / 1536, k = idx % 1536;
    zcs[b][k] = (k < DUNITS) ? h1[b * DUNITS + k] : att_c[b * 512 + (k - DUNITS)];
  }
  __syncthreads();
  if (tid < BDIM * 81) {
    int b = tid / 81, o = tid % 81;
    if (o < ODIM) {
      float acc = 0.f;
      const float* w = feat_w + (size_t)o * 1536;
      for (int k = 0; k < 1536; ++k) acc += zcs[b][k] * w[k];
      before_out[((size_t)b * LSTEPS + step) * ODIM + o] = acc;
      x0_bf[((size_t)b * ODIM + o) * LSTEPS + step] = (__bf16)acc;
    } else {
      float acc = prob_b[0];
      for (int k = 0; k < 1536; ++k) acc += zcs[b][k] * prob_w[k];
      logits[b * LSTEPS + step] = acc;
    }
  }
}

// ---------------------------------------------------------------------------
// postnet conv1d (k=5, pad 2) as im2col WMMA GEMM + fused BN (+tanh).
// Two 16-row t-tiles per block share each weight fragment (two independent
// WMMA chains); im2col tiles built in LDS in A-fragment layout.
// grid (B, 8 t-pairs, Cout/16), 32 threads.
// ---------------------------------------------------------------------------
__global__ void k_postconv(const __bf16* __restrict__ x, const __bf16* __restrict__ w,
                           int Cin, int Cout, int K, int KTiles,
                           const float* __restrict__ gamma, const float* __restrict__ beta,
                           const float* __restrict__ mean, const float* __restrict__ var,
                           int apply_tanh, int last, __bf16* y_bf,
                           const float* __restrict__ before, float* after) {
  const int L = LSTEPS;
  int b = blockIdx.x, t0 = blockIdx.y * 32, c0 = blockIdx.z * 16;
  __shared__ alignas(32) __bf16 atile[2][512];
  int lane = threadIdx.x;
  const __bf16* wt = w + (size_t)blockIdx.z * KTiles * 512;
  v8f acc0 = {0.f, 0.f, 0.f, 0.f, 0.f, 0.f, 0.f, 0.f};
  v8f acc1 = acc0;

  for (int kt = 0; kt < KTiles; ++kt) {
    int k0 = kt * 32;
    // fill both A tiles in fragment layout: idx = tsel*512 + l*16 + i
    for (int idx = lane; idx < 1024; idx += 32) {
      int tsel = idx >> 9;
      int r = idx & 511;
      int l = r >> 4, i = r & 15;
      int m = l & 15;
      int kb = (l & 16) ? 8 : 0;
      int kin = (i < 8) ? (kb + i) : (16 + kb + (i - 8));
      int k = k0 + kin;
      __bf16 v = (__bf16)0.f;
      if (k < K) {
        int ci = k / 5, dk = k % 5;
        int tt = t0 + tsel * 16 + m + dk - 2;
        if (tt >= 0 && tt < L) v = x[((size_t)b * Cin + ci) * L + tt];
      }
      atile[tsel][r] = v;
    }
    __syncthreads();
    v16bf bfrag = load_frag(wt + (size_t)kt * 512);
    acc0 = wmma_bf16(load_frag(atile[0]), bfrag, acc0);
    acc1 = wmma_bf16(load_frag(atile[1]), bfrag, acc1);
    __syncthreads();
  }

  int n = lane & 15, mb = (lane & 16) ? 8 : 0;
  int co = c0 + n;
  float sc = gamma[co] * rsqrtf(var[co] + 1e-5f);
  float sh = beta[co] - mean[co] * sc;
#pragma unroll
  for (int i = 0; i < 8; ++i) {
#pragma unroll
    for (int tsel = 0; tsel < 2; ++tsel) {
      int t = t0 + tsel * 16 + mb + i;
      float av = (tsel == 0) ? acc0[i] : acc1[i];
      if (t < L) {
        float v = av * sc + sh;
        if (apply_tanh) v = tanhf(v);
        if (last) {
          size_t oi = ((size_t)b * L + t) * ODIM + co;
          after[oi] = before[oi] + v;
        } else {
          y_bf[((size_t)b * Cout + co) * L + t] = (__bf16)v;
        }
      }
    }
  }
}

// ---------------------------------------------------------------------------
// host orchestration
// ---------------------------------------------------------------------------
extern "C" void kernel_launch(void* const* d_in, const int* in_sizes, int n_in,
                              void* d_out, int out_size, void* d_ws, size_t ws_size,
                              hipStream_t stream) {
  const float* hs       = (const float*)d_in[0];
  const int*   hlens    = (const int*)d_in[1];
  const float* trans    = (const float*)d_in[2];
  const float* ys       = (const float*)d_in[3];
  const float* W_enc    = (const float*)d_in[4];
  const float* b_enc    = (const float*)d_in[5];
  const float* W_dec    = (const float*)d_in[6];
  const float* W_att    = (const float*)d_in[7];
  const float* loc_conv = (const float*)d_in[8];
  const float* gvec_w   = (const float*)d_in[9];
  const float* gvec_b   = (const float*)d_in[10];
  const float* pre1_W = (const float*)d_in[11];
  const float* pre1_b = (const float*)d_in[12];
  const float* pre2_W = (const float*)d_in[13];
  const float* pre2_b = (const float*)d_in[14];
  const float* Wih0 = (const float*)d_in[15];
  const float* Whh0 = (const float*)d_in[16];
  const float* bih0 = (const float*)d_in[17];
  const float* bhh0 = (const float*)d_in[18];
  const float* Wih1 = (const float*)d_in[19];
  const float* Whh1 = (const float*)d_in[20];
  const float* bih1 = (const float*)d_in[21];
  const float* bhh1 = (const float*)d_in[22];
  const float* feat_out = (const float*)d_in[23];
  const float* prob_W   = (const float*)d_in[24];
  const float* prob_b   = (const float*)d_in[25];

  char* ws = (char*)d_ws;
  size_t off = 0;
  auto carve = [&](size_t bytes) { void* p = ws + off; off += (bytes + 255) & ~(size_t)255; return p; };
  __bf16* WIH0T = (__bf16*)carve((size_t)768 * 4096 * 2);
  __bf16* WHH0T = (__bf16*)carve((size_t)1024 * 4096 * 2);
  __bf16* WIH1T = (__bf16*)carve((size_t)1024 * 4096 * 2);
  __bf16* WHH1T = (__bf16*)carve((size_t)1024 * 4096 * 2);
  __bf16* WPOST[5];
  for (int l = 0; l < 5; ++l) WPOST[l] = (__bf16*)carve((size_t)2560 * 512 * 2);
  __bf16* HSWZ   = (__bf16*)carve((size_t)1024 * 512 * 2);
  __bf16* WENCWZ = (__bf16*)carve((size_t)512 * 256 * 2);
  __bf16* XSWZ   = (__bf16*)carve((size_t)16 * XSK * 2);
  __bf16* H0WZ   = (__bf16*)carve((size_t)16 * DUNITS * 2);
  __bf16* H1WZ   = (__bf16*)carve((size_t)16 * DUNITS * 2);
  __bf16* X0BF   = (__bf16*)carve((size_t)4 * 512 * LSTEPS * 2);
  __bf16* X1BF   = (__bf16*)carve((size_t)4 * 512 * LSTEPS * 2);
  float* PRE_ENC  = (float*)carve((size_t)1024 * 256 * 4);
  float* ATT_PREV = (float*)carve((size_t)BDIM * TDIM * 4);
  float* ATT_C    = (float*)carve((size_t)BDIM * 512 * 4);
  float* GATES    = (float*)carve((size_t)BDIM * 4096 * 4);
  float* H0 = (float*)carve((size_t)BDIM * DUNITS * 4);
  float* C0 = (float*)carve((size_t)BDIM * DUNITS * 4);
  float* H1 = (float*)carve((size_t)BDIM * DUNITS * 4);
  float* C1 = (float*)carve((size_t)BDIM * DUNITS * 4);

  float* out_after  = (float*)d_out;                       // (B,L,80)
  float* out_before = out_after + (size_t)BDIM * LSTEPS * ODIM;
  float* out_logits = out_before + (size_t)BDIM * LSTEPS * ODIM;
  float* out_attw   = out_logits + (size_t)BDIM * LSTEPS;

  // ---- prolog: init + fragment-swizzled bf16 weights + encoder projection ----
  k_init<<<64, 256, 0, stream>>>(H0, C0, H1, C1, XSWZ, H0WZ, H1WZ, ATT_PREV, hlens);
  k_wswz_nk<<<((4096 * 768) + 255) / 256, 256, 0, stream>>>(Wih0, WIH0T, 4096, 768);
  k_wswz_nk<<<((4096 * 1024) + 255) / 256, 256, 0, stream>>>(Whh0, WHH0T, 4096, 1024);
  k_wswz_nk<<<((4096 * 1024) + 255) / 256, 256, 0, stream>>>(Wih1, WIH1T, 4096, 1024);
  k_wswz_nk<<<((4096 * 1024) + 255) / 256, 256, 0, stream>>>(Whh1, WHH1T, 4096, 1024);
  k_hs_swz<<<(524288 + 255) / 256, 256, 0, stream>>>(hs, HSWZ);
  k_wswz_kn<<<(131072 + 255) / 256, 256, 0, stream>>>(W_enc, WENCWZ, 256, 512);
  const int cins[5]  = {80, 512, 512, 512, 512};
  const int couts[5] = {512, 512, 512, 512, 80};
  int kpads[5];
  for (int l = 0; l < 5; ++l) {
    const float* cw = (const float*)d_in[26 + 5 * l];
    int K = cins[l] * 5;
    kpads[l] = (K + 31) & ~31;
    size_t total = (size_t)(couts[l] / 16) * (kpads[l] / 32) * 512;
    k_pwswz<<<(total + 255) / 256, 256, 0, stream>>>(cw, WPOST[l], couts[l], cins[l]);
  }
  k_preenc<<<dim3(32, 16), 32, 0, stream>>>(HSWZ, WENCWZ, b_enc, PRE_ENC);

  // ---- sequential decode: 250 steps ----
  for (int t = 0; t < LSTEPS; ++t) {
    k_attention<<<BDIM, 256, 0, stream>>>(hs, hlens, trans, PRE_ENC, H0, W_dec, W_att,
                                          loc_conv, gvec_w, gvec_b, ATT_PREV, ATT_C,
                                          XSWZ, out_attw, t);
    k_prenet<<<1, 256, 0, stream>>>(ys, t, pre1_W, pre1_b, pre2_W, pre2_b, XSWZ);
    k_gates<24, 32><<<128, 32, 0, stream>>>(XSWZ, WIH0T, H0WZ, WHH0T,
                                            bih0, bhh0, GATES, 4096);
    k_lstm_update<<<16, 256, 0, stream>>>(GATES, H0, C0, H0WZ);
    k_gates<32, 32><<<128, 32, 0, stream>>>(H0WZ, WIH1T, H1WZ, WHH1T,
                                            bih1, bhh1, GATES, 4096);
    k_lstm_update<<<16, 256, 0, stream>>>(GATES, H1, C1, H1WZ);
    k_output<<<1, 512, 0, stream>>>(H1, ATT_C, feat_out, prob_W, prob_b,
                                    out_before, out_logits, X0BF, t);
  }

  // ---- postnet: 5 im2col WMMA convs, fused BN/tanh, residual into d_out ----
  __bf16* pin = X0BF;
  __bf16* pout = X1BF;
  for (int l = 0; l < 5; ++l) {
    const float* gamma = (const float*)d_in[26 + 5 * l + 1];
    const float* beta  = (const float*)d_in[26 + 5 * l + 2];
    const float* mean  = (const float*)d_in[26 + 5 * l + 3];
    const float* var   = (const float*)d_in[26 + 5 * l + 4];
    dim3 grid(BDIM, 8, couts[l] / 16);
    k_postconv<<<grid, 32, 0, stream>>>(pin, WPOST[l], cins[l], couts[l], cins[l] * 5,
                                        kpads[l] / 32, gamma, beta, mean, var,
                                        (l < 4) ? 1 : 0, (l == 4) ? 1 : 0,
                                        pout, out_before, out_after);
    __bf16* tmp = pin; pin = pout; pout = tmp;
  }
}
